// LSTMTSSparseToTTE_90220083020376
// MI455X (gfx1250) — compile-verified, matrix operand order checked by reading, not run on hardware
//
#include <hip/hip_runtime.h>
#include <hip/hip_bf16.h>

// ---------------------------------------------------------------------------
// LSTM (T=1024, B=256, E=128) + softplus, fused persistent-per-batch-tile
// kernel using v_wmma_f32_16x16x32_bf16 on gfx1250 (wave32).
//
// 16 workgroups (one 16-batch tile each) x 1024 threads (32 waves).
// Each wave owns ONE 16-wide gate-column tile. gfx1250 caps waves at 128
// VGPRs for 1024-thread blocks, so the 12 weight B-fragments are split:
// 9 register-resident (72 VGPRs) + 3 in a per-lane LDS stash (no barrier
// needed: each lane touches only its own slot). An empty asm on the stash
// offset defeats LICM so those ds_loads stay in the loop instead of being
// hoisted into (spilled) registers.
// ---------------------------------------------------------------------------

typedef __bf16 v16bf __attribute__((ext_vector_type(16)));
typedef __bf16 v8bf  __attribute__((ext_vector_type(8)));
typedef float  v8f   __attribute__((ext_vector_type(8)));

#define T_STEPS 1024
#define BATCH   256
#define HID     128   // E
#define INW     256   // 2E
#define GW      512   // 4E

// padded LDS strides (in elements) to spread ds_load_b128 across banks
#define XS 264   // xlds row stride (bf16), rows stay 16-byte aligned
#define HS 136   // hlds row stride (bf16)
#define GS 516   // gates row stride (f32)

#define NLDSF 3              // W_ih fragments kept in the LDS stash
#define FSTRIDE (32 * 16)    // halfs per (wave,frag) stash block: 32 lanes x 16

union FragU { v16bf v; v8bf h[2]; };

__device__ __forceinline__ v16bf mkfrag(const __bf16* lo, const __bf16* hi) {
    FragU u;
    u.h[0] = *(const v8bf*)lo;   // k = K0 + kb    .. +7  -> frag VGPRs 0..3
    u.h[1] = *(const v8bf*)hi;   // k = K0+16+kb  .. +7   -> frag VGPRs 4..7
    return u.v;
}

__device__ __forceinline__ float fsig(float x)  { return 1.0f / (1.0f + __expf(-x)); }
__device__ __forceinline__ float ftanh(float x) { return 2.0f * fsig(2.0f * x) - 1.0f; }
__device__ __forceinline__ float fsoftplus(float x) {
    return (x > 20.0f) ? x : __logf(1.0f + __expf(x));
}

// ---------------------------------------------------------------------------
// Pre-pass: fp32 weights -> bf16 in workspace, bias sum.
// ---------------------------------------------------------------------------
__global__ void lstm_prep_kernel(const float* __restrict__ W_ih,
                                 const float* __restrict__ W_hh,
                                 const float* __restrict__ b_ih,
                                 const float* __restrict__ b_hh,
                                 __bf16* __restrict__ wih_bf,
                                 __bf16* __restrict__ whh_bf,
                                 float*  __restrict__ bsum) {
    int i = blockIdx.x * blockDim.x + threadIdx.x;
    const int NW1 = GW * INW;          // 131072
    const int NW2 = GW * HID;          // 65536
    if (i < NW1) {
        wih_bf[i] = (__bf16)W_ih[i];
    } else if (i < NW1 + NW2) {
        int j = i - NW1;
        whh_bf[j] = (__bf16)W_hh[j];
    } else if (i < NW1 + NW2 + GW) {
        int j = i - NW1 - NW2;
        bsum[j] = b_ih[j] + b_hh[j];
    }
}

// ---------------------------------------------------------------------------
// Main fused LSTM kernel: grid = 16 blocks (one batch tile), 1024 threads.
// ---------------------------------------------------------------------------
__global__ __launch_bounds__(1024, 1)
void lstm_main_kernel(const float* __restrict__ x,
                      const float* __restrict__ h0,
                      const float* __restrict__ c0,
                      const __bf16* __restrict__ wih,   // [GW][INW] bf16 row-major
                      const __bf16* __restrict__ whh,   // [GW][HID] bf16 row-major
                      const float*  __restrict__ bsum,  // [GW]
                      float* __restrict__ out) {
    __shared__ __align__(16) __bf16 xlds[16 * XS];               // x_t tile, bf16
    __shared__ __align__(16) __bf16 hlds[16 * HS];               // h tile, bf16
    __shared__ __align__(16) float  gateslds[16 * GS];           // gates fp32
    __shared__ __align__(16) __bf16 blds[32 * NLDSF * FSTRIDE];  // weight stash

    const int tid  = threadIdx.x;
    const int b0   = blockIdx.x * 16;      // batch tile base
    const int lane = tid & 31;
    const int wv   = tid >> 5;             // wave 0..31 == gate-column tile
    const int m    = lane & 15;            // A row / B col / D col within tile
    const int kb   = (lane >> 4) << 3;     // 0 or 8 (half-wave K base)

    const int ct = wv;                     // this wave's gate-column tile
    const int n  = 16 * ct + m;            // global gate column (B frag row)
    const float bias = bsum[n];

    // ---- W_ih fragments 0..NLDSF-1 -> per-lane LDS stash (own slot only,
    //      written and read by the same lane => no barrier required) ----
#pragma unroll
    for (int f = 0; f < NLDSF; ++f) {
        const int K = 32 * f;
        v8bf lo = *(const v8bf*)&wih[(size_t)n * INW + K + kb];
        v8bf hi = *(const v8bf*)&wih[(size_t)n * INW + K + 16 + kb];
        int el = (wv * NLDSF + f) * FSTRIDE + lane * 16;
        *(v8bf*)&blds[el]     = lo;
        *(v8bf*)&blds[el + 8] = hi;
    }

    // ---- remaining weight B-fragments stay in registers (9 x 8 VGPRs) ----
    v16bf wihf[8 - NLDSF];
#pragma unroll
    for (int ks = NLDSF; ks < 8; ++ks) {
        const int K = 32 * ks;
        wihf[ks - NLDSF] = mkfrag(&wih[(size_t)n * INW + K + kb],
                                  &wih[(size_t)n * INW + K + 16 + kb]);
    }
    v16bf whhf[4];
#pragma unroll
    for (int ks = 0; ks < 4; ++ks) {
        const int K = 32 * ks;
        whhf[ks] = mkfrag(&whh[(size_t)n * HID + K + kb],
                          &whh[(size_t)n * HID + K + 16 + kb]);
    }

    // Elementwise ownership: 2 (b,e) pairs per thread, fixed for all t.
    float creg[2];
#pragma unroll
    for (int i = 0; i < 2; ++i) {
        int idx = tid + 1024 * i;
        int e = idx & (HID - 1);
        int b = idx >> 7;
        creg[i] = c0[(b0 + b) * HID + e];
        hlds[b * HS + e] = (__bf16)h0[(b0 + b) * HID + e];
    }

    // Prefetch x[0] tile (16 rows x 256 f32 = 1024 float4, rows contiguous).
    float4 pf;
    {
        const float4* xs = (const float4*)(x + ((size_t)0 * BATCH + b0) * INW);
        pf = xs[tid];
    }

    for (int t = 0; t < T_STEPS; ++t) {
        // ---- stage prefetched x tile into LDS as bf16 ----
        {
            int fi = tid;                  // float4 index 0..1023
            __bf16* d = &xlds[(fi >> 6) * XS + ((fi & 63) << 2)];
            d[0] = (__bf16)pf.x; d[1] = (__bf16)pf.y;
            d[2] = (__bf16)pf.z; d[3] = (__bf16)pf.w;
        }
        __syncthreads();   // barrier A: xlds + hlds(prev step) visible

        // ---- prefetch x[t+1] (latency hidden behind this step's WMMAs) ----
        {
            int tn = (t + 1 < T_STEPS) ? (t + 1) : t;
            const float4* xs = (const float4*)(x + ((size_t)tn * BATCH + b0) * INW);
            pf = xs[tid];
        }

        // ---- gates = x_t @ W_ih^T + h @ W_hh^T  (bias added at scatter) ----
        v8f acc = {};   // inline-0 C operand on the first WMMA

        // opaque base offset: keeps the stash ds_loads inside the loop
        int bel = wv * NLDSF * FSTRIDE + lane * 16;
        asm volatile("" : "+v"(bel));

#pragma unroll
        for (int ks = 0; ks < NLDSF; ++ks) {           // x part, stashed frags
            const int K = 32 * ks;
            v16bf a = mkfrag(&xlds[m * XS + K + kb],
                             &xlds[m * XS + K + 16 + kb]);
            v16bf bw = mkfrag(&blds[bel + ks * FSTRIDE],
                              &blds[bel + ks * FSTRIDE + 8]);
            acc = __builtin_amdgcn_wmma_f32_16x16x32_bf16(
                      false, a, false, bw, (short)0, acc, false, false);
        }
#pragma unroll
        for (int ks = NLDSF; ks < 8; ++ks) {           // x part, register frags
            const int K = 32 * ks;
            v16bf a = mkfrag(&xlds[m * XS + K + kb],
                             &xlds[m * XS + K + 16 + kb]);
            acc = __builtin_amdgcn_wmma_f32_16x16x32_bf16(
                      false, a, false, wihf[ks - NLDSF], (short)0, acc, false, false);
        }
#pragma unroll
        for (int ks = 0; ks < 4; ++ks) {               // h part: K = 128
            const int K = 32 * ks;
            v16bf a = mkfrag(&hlds[m * HS + K + kb],
                             &hlds[m * HS + K + 16 + kb]);
            acc = __builtin_amdgcn_wmma_f32_16x16x32_bf16(
                      false, a, false, whhf[ks], (short)0, acc, false, false);
        }

        // ---- scatter accumulator (+bias) to gates LDS (M = r+kb, N = m) ----
#pragma unroll
        for (int r = 0; r < 8; ++r) {
            gateslds[(r + kb) * GS + 16 * ct + m] = acc[r] + bias;
        }
        __syncthreads();   // barrier B: gates visible

        // ---- elementwise: c,h update + softplus output ----
#pragma unroll
        for (int i = 0; i < 2; ++i) {
            int idx = tid + 1024 * i;
            int e = idx & (HID - 1);
            int b = idx >> 7;
            float ig = gateslds[b * GS + e];
            float fg = gateslds[b * GS + HID + e];
            float gg = gateslds[b * GS + 2 * HID + e];
            float og = gateslds[b * GS + 3 * HID + e];
            float iv = fsig(ig);
            float fv = fsig(fg);
            float gv = ftanh(gg);
            float ov = fsig(og);
            float c = fv * creg[i] + iv * gv;
            creg[i] = c;
            float h = ov * ftanh(c);
            out[((size_t)t * BATCH + b0 + b) * HID + e] = fsoftplus(h);
            hlds[b * HS + e] = (__bf16)h;
        }
        // No barrier needed here: next staging touches only xlds, whose last
        // readers are separated by barrier B; hlds/gates hazards are covered
        // by barrier A of the next iteration.
    }
}

// ---------------------------------------------------------------------------
extern "C" void kernel_launch(void* const* d_in, const int* in_sizes, int n_in,
                              void* d_out, int out_size, void* d_ws, size_t ws_size,
                              hipStream_t stream) {
    (void)in_sizes; (void)n_in; (void)out_size; (void)ws_size;

    const float* x    = (const float*)d_in[0];  // [1024,256,256]
    const float* h0   = (const float*)d_in[1];  // [256,128]
    const float* c0   = (const float*)d_in[2];  // [256,128]
    const float* W_ih = (const float*)d_in[3];  // [512,256]
    const float* W_hh = (const float*)d_in[4];  // [512,128]
    const float* b_ih = (const float*)d_in[5];  // [512]
    const float* b_hh = (const float*)d_in[6];  // [512]
    float* out = (float*)d_out;                 // [1024,256,128]

    // workspace layout: wih_bf (256KB) | whh_bf (128KB) | bsum (2KB)
    char* ws = (char*)d_ws;
    __bf16* wih_bf = (__bf16*)ws;
    __bf16* whh_bf = (__bf16*)(ws + (size_t)GW * INW * sizeof(__bf16));
    float*  bsum   = (float*) (ws + (size_t)GW * INW * sizeof(__bf16)
                                  + (size_t)GW * HID * sizeof(__bf16));

    const int total = GW * INW + GW * HID + GW;
    lstm_prep_kernel<<<(total + 255) / 256, 256, 0, stream>>>(
        W_ih, W_hh, b_ih, b_hh, wih_bf, whh_bf, bsum);

    lstm_main_kernel<<<16, 1024, 0, stream>>>(
        x, h0, c0, wih_bf, whh_bf, bsum, out);
}
